// KmeansQuantizerInjector_43542378447256
// MI455X (gfx1250) — matrix operand order repaired and checked by reading, not compiled, not run on hardware
//
#include <hip/hip_runtime.h>

typedef float  v8f   __attribute__((ext_vector_type(8)));
typedef __bf16 v8bf  __attribute__((ext_vector_type(8)));
typedef __bf16 v16bf __attribute__((ext_vector_type(16)));

#define NAN_FILL_VAL 9999999999.0f

// Problem dims (match reference)
#define BDIM 16
#define CDIM 512          // K of the GEMM
#define SDIM 2048
#define NKC  8192         // number of centroids (N of the GEMM)
#define MTOT (BDIM * SDIM)

// Tiling: block = 128(M) x 128(N); 8 waves = 4(M) x 2(N); wave tile = 32x64 (2 mi x 4 ni)
#define MBLOCK 128
#define NBLOCK 128
#define NI 4
#define NKTILES (NKC / NBLOCK)       // 64
#define LDS_ROW_BF 520               // 512 + 8 pad (bf16 elems)
#define LDS_ROW_B  (LDS_ROW_BF * 2)  // 1040 bytes; 16B aligned; == 16 mod 256 -> conflict-free b128
#define SUBTILE_B  (16 * LDS_ROW_B)  // 16640 bytes between 16-row subtiles (fits DS imm offset)

// LDS layout (bytes)
#define XS_OFF   0
#define XS_SZ    (MBLOCK * LDS_ROW_B)      // 133120
#define CT_OFF   (XS_OFF + XS_SZ)
#define CT_SZ    (NBLOCK * LDS_ROW_B)      // 133120
#define CSQ_OFF  (CT_OFF + CT_SZ)
#define CSQ_SZ   (2 * NBLOCK * 4)          // double-buffered
#define REDV_OFF (CSQ_OFF + CSQ_SZ)
#define REDV_SZ  (MBLOCK * 2 * 4)
#define REDI_OFF (REDV_OFF + REDV_SZ)
#define REDI_SZ  (MBLOCK * 2 * 4)
#define LDS_TOTAL (REDI_OFF + REDI_SZ)     // 269312 bytes (< 320 KB / WGP)

__launch_bounds__(256)
__global__ void kmeans_label_kernel(const float* __restrict__ x,
                                    const float* __restrict__ cent,
                                    int* __restrict__ out) {
    extern __shared__ char smem[];
    char*  xsB  = smem + XS_OFF;   // bf16 [MBLOCK][LDS_ROW_BF]
    char*  ctB  = smem + CT_OFF;   // bf16 [NBLOCK][LDS_ROW_BF]
    float* csqA = (float*)(smem + CSQ_OFF);            // buffer 0
    float* csqB = (float*)(smem + CSQ_OFF) + NBLOCK;   // buffer 1
    float* redV = (float*)(smem + REDV_OFF);
    int*   redI = (int*)(smem + REDI_OFF);

    const int t     = threadIdx.x;
    const int lane  = t & 31;
    const int wave  = t >> 5;
    const int waveM = wave & 3;   // 4 waves along M (32 rows each)
    const int waveN = wave >> 2;  // 2 waves along N (64 cols each)
    const int l16   = lane & 15;
    const int half  = lane >> 4;

    const int blk = blockIdx.x;
    const int bi  = blk >> 4;              // 16 M-blocks per batch (2048/128)
    const int si0 = (blk & 15) * MBLOCK;

    // ---------------- Stage X block (fp32 -> bf16) into LDS, full K=512 ----------------
    {
        const int m = t & 127;
        const int g = t >> 7;   // 0..1
        const float* xb = x + (size_t)bi * (CDIM * SDIM) + si0 + m;
#pragma unroll 4
        for (int it = 0; it < 32; ++it) {
            const int chBase = (it * 2 + g) * 8;
            v8f v;
#pragma unroll
            for (int j = 0; j < 8; ++j)
                v[j] = xb[(size_t)(chBase + j) * SDIM];
            v8bf bv = __builtin_convertvector(v, v8bf);
            *(v8bf*)(xsB + m * LDS_ROW_B + chBase * 2) = bv;
        }
    }
    if (t < NBLOCK) csqA[t] = 0.0f;   // buffer for tile 0
    __syncthreads();

    // Running per-row argmin state: 2 M-subtiles x 8 accumulator rows per lane
    float minv[2][8];
    int   mini[2][8];
#pragma unroll
    for (int mi = 0; mi < 2; ++mi)
#pragma unroll
        for (int r = 0; r < 8; ++r) { minv[mi][r] = 3.0e38f; mini[mi][r] = 0; }

    // Single per-lane base pointers; all per-step displacements are DS immediates.
    // A (contiguous k = 16*half + s, same as B): lanes>=16 start at K=16 -> +32B
    // B (contiguous k = 16*half + s, same as A): lanes>=16 start at K=16 -> +32B
    const char* aBase = xsB + (waveM * 32 + l16) * LDS_ROW_B + half * 32;
    const char* bBase = ctB + (waveN * 64 + l16) * LDS_ROW_B + half * 32;

    for (int tile = 0; tile < NKTILES; ++tile) {
        const int kBase = tile * NBLOCK;
        float* csqCur = (tile & 1) ? csqB : csqA;
        float* csqNxt = (tile & 1) ? csqA : csqB;

        // ---- Load C tile: fp32 [512, 128] slice -> bf16 LDS [128][512], + csq atomics ----
        if (t < NBLOCK) csqNxt[t] = 0.0f;   // prep next tile's buffer (prev readers done at last barrier)
        {
            const int n  = t & 127;
            const int cg = t >> 7;   // 0..1 channel groups
            const float* cb = cent + kBase + n;
            float part = 0.0f;
#pragma unroll 2
            for (int it = 0; it < 32; ++it) {
                const int chBase = it * 16 + cg * 8;
                v8f v;
#pragma unroll
                for (int j = 0; j < 8; ++j)
                    v[j] = cb[(size_t)(chBase + j) * NKC];
#pragma unroll
                for (int j = 0; j < 8; ++j)
                    part += v[j] * v[j];
                v8bf bv = __builtin_convertvector(v, v8bf);
                *(v8bf*)(ctB + n * LDS_ROW_B + chBase * 2) = bv;
            }
            atomicAdd(&csqCur[n], part);   // ds_add_f32
        }
        // Prefetch next C tile into L2 (gfx1250 global_prefetch_b8); warms during WMMA burst
        if (tile + 1 < NKTILES) {
            const float* p = cent + (size_t)(t * 2) * NKC + (kBase + NBLOCK);
            __builtin_prefetch(p, 0, 1);
            __builtin_prefetch(p + 64, 0, 1);
            __builtin_prefetch(p + NKC, 0, 1);
            __builtin_prefetch(p + NKC + 64, 0, 1);
        }
        __syncthreads();

        // ---- WMMA compute: 8 x (16x16) f32 accumulators per wave, K=512 in 16 steps ----
        v8f acc[2][NI];
#pragma unroll
        for (int mi = 0; mi < 2; ++mi)
#pragma unroll
            for (int ni = 0; ni < NI; ++ni)
#pragma unroll
                for (int e = 0; e < 8; ++e)
                    acc[mi][ni][e] = 0.0f;

#pragma unroll
        for (int ks = 0; ks < 16; ++ks) {
            const int kb = ks * 64;   // 32 bf16 = 64 bytes per K-step
            v16bf a[2], b[NI];
#pragma unroll
            for (int mi = 0; mi < 2; ++mi) {
                v8bf lo = *(const v8bf*)(aBase + mi * SUBTILE_B + kb);
                v8bf hi = *(const v8bf*)(aBase + mi * SUBTILE_B + kb + 16);
                a[mi] = __builtin_shufflevector(lo, hi, 0,1,2,3,4,5,6,7,8,9,10,11,12,13,14,15);
            }
#pragma unroll
            for (int ni = 0; ni < NI; ++ni) {
                v8bf lo = *(const v8bf*)(bBase + ni * SUBTILE_B + kb);
                v8bf hi = *(const v8bf*)(bBase + ni * SUBTILE_B + kb + 16);
                b[ni] = __builtin_shufflevector(lo, hi, 0,1,2,3,4,5,6,7,8,9,10,11,12,13,14,15);
            }
#pragma unroll
            for (int mi = 0; mi < 2; ++mi)
#pragma unroll
                for (int ni = 0; ni < NI; ++ni)
                    acc[mi][ni] = __builtin_amdgcn_wmma_f32_16x16x32_bf16(
                        false, a[mi], false, b[ni], (short)0, acc[mi][ni], false, false);
        }

        // ---- Fused epilogue: dist = csq[n] - 2*dot ; running argmin ----
#pragma unroll
        for (int ni = 0; ni < NI; ++ni) {
            const int nl = waveN * 64 + ni * 16 + l16;   // D-matrix column = lane%16
            const float cs = csqCur[nl];
            const int   ng = kBase + nl;
#pragma unroll
            for (int mi = 0; mi < 2; ++mi) {
#pragma unroll
                for (int r = 0; r < 8; ++r) {
                    float d = cs - 2.0f * acc[mi][ni][r];
                    d = (d == d) ? d : NAN_FILL_VAL;     // NaN fill
                    if (d < minv[mi][r]) { minv[mi][r] = d; mini[mi][r] = ng; }
                }
            }
        }
        __syncthreads();   // Ct / csqCur consumed; safe to overwrite next iteration
    }

    // ---- Final reduction: butterfly over 16-lane column groups, combine the 2 N-waves ----
#pragma unroll
    for (int mi = 0; mi < 2; ++mi) {
#pragma unroll
        for (int r = 0; r < 8; ++r) {
            float v = minv[mi][r];
            int   i = mini[mi][r];
#pragma unroll
            for (int off = 8; off > 0; off >>= 1) {
                float v2 = __shfl_xor(v, off, 32);
                int   i2 = __shfl_xor(i, off, 32);
                if (v2 < v || (v2 == v && i2 < i)) { v = v2; i = i2; }
            }
            if (l16 == 0) {
                // D layout: VGPR r holds M=r (lanes 0-15) and M=r+8 (lanes 16-31)
                const int row = waveM * 32 + mi * 16 + r + half * 8;
                redV[row * 2 + waveN] = v;
                redI[row * 2 + waveN] = i;
            }
        }
    }
    __syncthreads();
    if (t < MBLOCK) {
        const float v0 = redV[t * 2 + 0], v1 = redV[t * 2 + 1];
        const int   i0 = redI[t * 2 + 0], i1 = redI[t * 2 + 1];
        const int idx = (v1 < v0 || (v1 == v0 && i1 < i0)) ? i1 : i0;
        out[(size_t)bi * SDIM + si0 + t] = idx;   // labels[b, s] row-major == m_global
    }
}

extern "C" void kernel_launch(void* const* d_in, const int* in_sizes, int n_in,
                              void* d_out, int out_size, void* d_ws, size_t ws_size,
                              hipStream_t stream) {
    (void)in_sizes; (void)n_in; (void)d_ws; (void)ws_size; (void)out_size;
    const float* x    = (const float*)d_in[0];   // [16, 512, 2048] fp32
    const float* cent = (const float*)d_in[1];   // [512, 8192] fp32
    int* out = (int*)d_out;                      // [16, 2048] int32 labels

    dim3 grid(MTOT / MBLOCK);   // 256 workgroups
    dim3 block(256);            // 8 wave32 waves
    hipLaunchKernelGGL(kmeans_label_kernel, grid, block, LDS_TOTAL, stream, x, cent, out);
}
